// DecoderBlock_15891378995289
// MI455X (gfx1250) — matrix operand order repaired and checked by reading, not compiled, not run on hardware
//
#include <hip/hip_runtime.h>
#include <hip/hip_bf16.h>

typedef __attribute__((ext_vector_type(16))) _Float16 v16h;
typedef __attribute__((ext_vector_type(8)))  float    v8f;
typedef __attribute__((ext_vector_type(4)))  float    vf4;

namespace {

constexpr int BM  = 128;
constexpr int BN  = 128;
constexpr int BKK = 32;
constexpr int LDT = 48;   // padded LDS row stride (f16 elems) -> 96B, 16B-aligned rows

enum Epi { EPI_F32 = 0, EPI_SIG, EPI_RESID_F16, EPI_BIAS_F16, EPI_GBN_F16, EPI_GBN_F32, EPI_UP };

__device__ __forceinline__ float gelu_exact(float x) {
  return 0.5f * x * (1.0f + erff(x * 0.70710678118654752f));
}

// ---------------------------------------------------------------------------
// Tiled WMMA GEMM:  Out[M x 4096] (+epilogue) = A[M x K] (f16) * X[K x 4096] (f16)
// Double-buffered LDS pipeline; A tile staged with async global->LDS copies
// (ASYNCcnt), B tile staged with a register transpose.
// CONV=1: implicit GEMM for grouped 3x3 conv, K = 9*256, with shifted B loads.
// N is compile-time 4096 so epilogue stores fold into immediate offsets.
// ---------------------------------------------------------------------------
template<int CONV, int EPI>
__global__ __launch_bounds__(256)
void gemm_wmma_kernel(const _Float16* __restrict__ A,
                      const _Float16* __restrict__ X,
                      float* __restrict__ outF,
                      _Float16* __restrict__ outH,
                      const float* __restrict__ bias,
                      const float* __restrict__ bng,
                      const float* __restrict__ bnb,
                      const float* __restrict__ resid,
                      int K,
                      long strideXb, long strideOb, long strideRb,
                      int chanOff)
{
  constexpr int N = 4096;             // H*W, fixed for this problem
  __shared__ _Float16 As[2][BM * LDT];
  __shared__ _Float16 Bs[2][BN * LDT];

  const int tid  = threadIdx.x;
  const int lane = tid & 31;
  const int wid  = tid >> 5;
  const int wm   = (wid & 3) * 32;    // wave M offset within tile
  const int wn   = (wid >> 2) * 64;   // wave N offset within tile
  const int b    = blockIdx.z;
  const int m0   = blockIdx.y * BM;
  const int n0   = blockIdx.x * BN;
  const _Float16* Xb = X + (long)b * strideXb;

  v8f acc[2][4] = {};

  // staging index maps (256 threads)
  const int ar = tid >> 1;            // A row 0..127
  const int ac = (tid & 1) * 16;      // A col 0 or 16
  const int bk = tid >> 3;            // B k-row 0..31
  const int bn = (tid & 7) * 16;      // B n base 0..112
  const int cbase = CONV ? ((m0 >= 256) ? 256 : 0) : 0;

  // async copy of this thread's 32B A-slice into LDS buffer `buf`
  auto stageA = [&](int buf, int k0) {
    const _Float16* src = A + (long)(m0 + ar) * K + (k0 + ac);
    const unsigned lds0 = (unsigned)(size_t)(&As[buf][ar * LDT + ac]);
    const unsigned long long ga = (unsigned long long)(size_t)src;
    asm volatile("global_load_async_to_lds_b128 %0, %1, off"
                 :: "v"(lds0), "v"(ga) : "memory");
    asm volatile("global_load_async_to_lds_b128 %0, %1, off"
                 :: "v"(lds0 + 16u), "v"(ga + 16ull) : "memory");
    if (k0 + BKK < K) __builtin_prefetch(src + BKK, 0, 1);
  };

  auto stageB = [&](int buf, int k0) {
    if constexpr (!CONV) {
      const _Float16* src = Xb + (long)(k0 + bk) * N + (n0 + bn);
      union { vf4 f[2]; _Float16 h[16]; } u;
      u.f[0] = *(const vf4*)(src);
      u.f[1] = *(const vf4*)(src + 8);
      #pragma unroll
      for (int j = 0; j < 16; ++j)
        Bs[buf][(bn + j) * LDT + bk] = u.h[j];
      if (k0 + BKK < K) __builtin_prefetch(src + (long)BKK * N, 0, 1);
    } else {
      const int kg  = k0 + bk;        // 0..2303
      const int tap = kg >> 8;        // 0..8
      const int c   = kg & 255;
      const int dy  = tap / 3 - 1;
      const int dx  = tap % 3 - 1;
      const _Float16* chan = Xb + (long)(cbase + c) * N;
      const int nb = n0 + bn;
      const int hh = (nb >> 6) + dy;  // all 16 pixels share one image row
      const int wb = nb & 63;
      if (dx == 0 && (unsigned)hh < 64u) {
        // interior tap: contiguous, 32B-aligned -> vector loads
        const _Float16* src = chan + hh * 64 + wb;
        union { vf4 f[2]; _Float16 h[16]; } u;
        u.f[0] = *(const vf4*)(src);
        u.f[1] = *(const vf4*)(src + 8);
        #pragma unroll
        for (int j = 0; j < 16; ++j)
          Bs[buf][(bn + j) * LDT + bk] = u.h[j];
      } else {
        #pragma unroll
        for (int j = 0; j < 16; ++j) {
          const int w = wb + j + dx;
          _Float16 val = (_Float16)0.0f;
          if ((unsigned)hh < 64u && (unsigned)w < 64u) val = chan[hh * 64 + w];
          Bs[buf][(bn + j) * LDT + bk] = val;
        }
      }
    }
  };

  // prologue: stage first tiles
  stageA(0, 0);
  stageB(0, 0);
  asm volatile("s_wait_asynccnt 0x0" ::: "memory");
  __syncthreads();

  const int hsel = lane >> 4;         // 0/1
  const int rsel = lane & 15;
  int buf = 0;

  for (int k0 = 0; k0 < K; k0 += BKK) {
    const int nk = k0 + BKK;
    const bool more = nk < K;
    if (more) {                        // stage next tile while computing
      stageA(buf ^ 1, nk);
      stageB(buf ^ 1, nk);
    }

    // ---- fragments per documented 16-bit WMMA lane layouts ----
    v16h afr[2], bfr[4];
    #pragma unroll
    for (int i = 0; i < 2; ++i) {
      const _Float16* p = &As[buf][(wm + i * 16 + rsel) * LDT];
      union { vf4 f[2]; v16h v; } u;
      u.f[0] = *(const vf4*)(p + 8 * hsel);        // K = 8h .. 8h+7
      u.f[1] = *(const vf4*)(p + 16 + 8 * hsel);   // K = 16+8h .. 16+8h+7
      afr[i] = u.v;
    }
    #pragma unroll
    for (int j = 0; j < 4; ++j) {
      const _Float16* p = &Bs[buf][(wn + j * 16 + rsel) * LDT];
      union { vf4 f[2]; v16h v; } u;
      u.f[0] = *(const vf4*)(p + 8 * hsel);        // K = 8h .. 8h+7
      u.f[1] = *(const vf4*)(p + 16 + 8 * hsel);   // K = 16+8h .. 16+8h+7
      bfr[j] = u.v;
    }
    #pragma unroll
    for (int i = 0; i < 2; ++i)
      #pragma unroll
      for (int j = 0; j < 4; ++j)
        acc[i][j] = __builtin_amdgcn_wmma_f32_16x16x32_f16(
            false, afr[i], false, bfr[j], (short)0, acc[i][j], false, false);

    if (more) asm volatile("s_wait_asynccnt 0x0" ::: "memory");
    __syncthreads();
    buf ^= 1;
  }

  // ---- epilogue ----
  // C/D layout: lane col n = lane&15, VGPR r -> row m = r + 8*(lane>>4).
  // One 64-bit base per 16x16 tile; per-r offsets are compile-time constants
  // (N = 4096) so stores fold into immediate instruction offsets.
  const int nl = lane & 15;
  const int mh = (lane >> 4) * 8;
  #pragma unroll
  for (int i = 0; i < 2; ++i) {
    #pragma unroll
    for (int j = 0; j < 4; ++j) {
      const int mb = m0 + wm + i * 16 + mh;          // row base (multiple of 8)
      const int n  = n0 + wn + j * 16 + nl;
      if constexpr (EPI == EPI_UP) {
        const int h = n >> 6, w = n & 63;
        float* po = outF + (((long)b * 128 + (mb >> 2)) * 128 + h * 2) * 128 + w * 2;
        #pragma unroll
        for (int r = 0; r < 8; ++r) {
          const float val = acc[i][j][r] + bias[mb + r];
          po[(r >> 2) * 16384 + ((r >> 1) & 1) * 128 + (r & 1)] = val;
        }
      } else {
        const long ob = (long)b * strideOb + (long)(chanOff + mb) * N + n;
        const long rb = (EPI == EPI_RESID_F16)
                          ? ((long)b * strideRb + (long)mb * N + n) : 0;
        #pragma unroll
        for (int r = 0; r < 8; ++r) {
          float val = acc[i][j][r];
          if constexpr (EPI == EPI_F32) {
            outF[ob + r * N] = val;
          } else if constexpr (EPI == EPI_SIG) {
            outF[ob + r * N] = 1.0f / (1.0f + __expf(-val));
          } else if constexpr (EPI == EPI_RESID_F16) {
            val += resid[rb + r * N];
            outH[ob + r * N] = (_Float16)val;
          } else if constexpr (EPI == EPI_BIAS_F16) {
            val += bias[mb + r];
            outH[ob + r * N] = (_Float16)val;
          } else if constexpr (EPI == EPI_GBN_F16 || EPI == EPI_GBN_F32) {
            val = gelu_exact(val + bias[mb + r]);
            val = val * 0.9999950000374997f * bng[mb + r] + bnb[mb + r];  // bn_eval
            if constexpr (EPI == EPI_GBN_F16)
              outH[ob + r * N] = (_Float16)val;
            else
              outF[ob + r * N] = val;
          }
        }
      }
    }
  }
}

// ---------------------------------------------------------------------------
// LayerNorm over channels (C=256), one thread per pixel, two coalesced passes.
// ---------------------------------------------------------------------------
template<int OUT16>
__global__ __launch_bounds__(256)
void ln_chan_kernel(const float* __restrict__ x,
                    const float* __restrict__ w, const float* __restrict__ bia,
                    float* __restrict__ outF, _Float16* __restrict__ outH)
{
  const int p = blockIdx.x * 256 + threadIdx.x;   // 0..32767
  const int b = p >> 12;
  const int s = p & 4095;
  const long base = (long)b * (256L * 4096) + s;
  float sum = 0.f, sq = 0.f;
  for (int c = 0; c < 256; ++c) {
    float v = x[base + (long)c * 4096];
    sum += v; sq += v * v;
  }
  const float mu  = sum * (1.0f / 256.0f);
  const float var = sq * (1.0f / 256.0f) - mu * mu;
  const float inv = rsqrtf(var + 1e-5f);
  for (int c = 0; c < 256; ++c) {
    float v = x[base + (long)c * 4096];
    float r = (v - mu) * inv * w[c] + bia[c];
    if (OUT16) outH[base + (long)c * 4096] = (_Float16)r;
    else       outF[base + (long)c * 4096] = r;
  }
}

// combine: wkv = 0.5*(h+v); LN(kn); *= sr; -> f16
__global__ __launch_bounds__(256)
void combine_kernel(const float* __restrict__ wh, const float* __restrict__ wv,
                    const float* __restrict__ sr,
                    const float* __restrict__ knw, const float* __restrict__ knb,
                    _Float16* __restrict__ out)
{
  const int p = blockIdx.x * 256 + threadIdx.x;
  const int b = p >> 12;
  const int s = p & 4095;
  const long base = (long)b * (256L * 4096) + s;
  float sum = 0.f, sq = 0.f;
  for (int c = 0; c < 256; ++c) {
    const long idx = base + (long)c * 4096;
    float v = 0.5f * (wh[idx] + wv[idx]);
    sum += v; sq += v * v;
  }
  const float mu  = sum * (1.0f / 256.0f);
  const float var = sq * (1.0f / 256.0f) - mu * mu;
  const float inv = rsqrtf(var + 1e-5f);
  for (int c = 0; c < 256; ++c) {
    const long idx = base + (long)c * 4096;
    float v = 0.5f * (wh[idx] + wv[idx]);
    float r = (v - mu) * inv * knw[c] + knb[c];
    out[idx] = (_Float16)(r * sr[idx]);
  }
}

// q_shift + token mixes -> f16 xk/xv/xr
__global__ __launch_bounds__(256)
void qshift_mix_kernel(const float* __restrict__ xn, const float* __restrict__ mask,
                       const float* __restrict__ mk, const float* __restrict__ mv,
                       const float* __restrict__ mr,
                       _Float16* __restrict__ xk, _Float16* __restrict__ xv,
                       _Float16* __restrict__ xr)
{
  const long i = (long)blockIdx.x * 256 + threadIdx.x;   // < 8*256*4096
  const int w = (int)(i & 63);
  const int h = (int)((i >> 6) & 63);
  const int c = (int)((i >> 12) & 255);
  const long b = i >> 20;
  const float xc = xn[i];
  const int quarter = c >> 6;
  float xs = 0.f;
  if      (quarter == 0) { if (w > 0)  xs = xn[i - 1];  }
  else if (quarter == 1) { if (w < 63) xs = xn[i + 1];  }
  else if (quarter == 2) { if (h > 0)  xs = xn[i - 64]; }
  else                   { if (h < 63) xs = xn[i + 64]; }
  const float m = mask[b * 4096 + h * 64 + w];
  const float xsh = m * xs + (1.f - m) * xc;
  const float k = mk[c], v = mv[c], r = mr[c];
  xk[i] = (_Float16)(xc * k + xsh * (1.f - k));
  xv[i] = (_Float16)(xc * v + xsh * (1.f - v));
  xr[i] = (_Float16)(xc * r + xsh * (1.f - r));
}

// WKV numerically-stable scan, horizontal (along W). thread = (b,c,h), lanes vary h.
__global__ __launch_bounds__(256)
void wkv_hscan_kernel(const float* __restrict__ k, const float* __restrict__ v,
                      const float* __restrict__ decay, const float* __restrict__ first,
                      float* __restrict__ out)
{
  const int t = blockIdx.x * 256 + threadIdx.x;   // < 8*256*64
  const int h = t & 63;
  const int c = (t >> 6) & 255;
  const int b = t >> 14;
  const long base = (((long)b * 256 + c) * 64 + h) * 64;
  const float wn = -__expf(decay[c]);
  const float u  = first[c];
  float a = 0.f, bb = 0.f, p = -1e38f;
  for (int w = 0; w < 64; ++w) {
    const float kt = k[base + w], vt = v[base + w];
    const float q  = fmaxf(p, u + kt);
    const float e1 = __expf(p - q), e2 = __expf(u + kt - q);
    out[base + w]  = (e1 * a + e2 * vt) / (e1 * bb + e2);
    const float q2 = fmaxf(p + wn, kt);
    const float f1 = __expf(p + wn - q2), f2 = __expf(kt - q2);
    a = f1 * a + f2 * vt; bb = f1 * bb + f2; p = q2;
  }
}

// vertical scan (along H). thread = (b,c,w), lanes vary w -> coalesced.
__global__ __launch_bounds__(256)
void wkv_vscan_kernel(const float* __restrict__ k, const float* __restrict__ v,
                      const float* __restrict__ decay, const float* __restrict__ first,
                      float* __restrict__ out)
{
  const int t = blockIdx.x * 256 + threadIdx.x;   // < 8*256*64
  const int w = t & 63;
  const int c = (t >> 6) & 255;
  const int b = t >> 14;
  const long base = ((long)b * 256 + c) * 4096 + w;
  const float wn = -__expf(decay[c]);
  const float u  = first[c];
  float a = 0.f, bb = 0.f, p = -1e38f;
  for (int h = 0; h < 64; ++h) {
    const long idx = base + (long)h * 64;
    const float kt = k[idx], vt = v[idx];
    const float q  = fmaxf(p, u + kt);
    const float e1 = __expf(p - q), e2 = __expf(u + kt - q);
    out[idx]       = (e1 * a + e2 * vt) / (e1 * bb + e2);
    const float q2 = fmaxf(p + wn, kt);
    const float f1 = __expf(p + wn - q2), f2 = __expf(kt - q2);
    a = f1 * a + f2 * vt; bb = f1 * bb + f2; p = q2;
  }
}

__global__ __launch_bounds__(256)
void f32_to_f16_kernel(const float* __restrict__ in, _Float16* __restrict__ out, long n)
{
  const long i = (long)blockIdx.x * 256 + threadIdx.x;
  if (i < n) out[i] = (_Float16)in[i];
}

// c1_w (512,256,3,3) -> f16 [o][tap*256 + c]
__global__ __launch_bounds__(256)
void c1_perm_kernel(const float* __restrict__ w, _Float16* __restrict__ out)
{
  const long i = (long)blockIdx.x * 256 + threadIdx.x;
  if (i < 512L * 2304) {
    const int o = (int)(i / 2304);
    const int r = (int)(i % 2304);
    const int t = r >> 8;
    const int c = r & 255;
    out[i] = (_Float16)w[((long)o * 256 + c) * 9 + t];
  }
}

} // namespace

extern "C" void kernel_launch(void* const* d_in, const int* in_sizes, int n_in,
                              void* d_out, int out_size, void* d_ws, size_t ws_size,
                              hipStream_t stream)
{
  (void)in_sizes; (void)n_in; (void)out_size; (void)ws_size;
  const float* x      = (const float*)d_in[0];
  const float* skip   = (const float*)d_in[1];
  const float* mask   = (const float*)d_in[2];
  const float* ln1_w  = (const float*)d_in[3];
  const float* ln1_b  = (const float*)d_in[4];
  const float* ln2_w  = (const float*)d_in[5];
  const float* ln2_b  = (const float*)d_in[6];
  const float* mix_k  = (const float*)d_in[7];
  const float* mix_v  = (const float*)d_in[8];
  const float* mix_r  = (const float*)d_in[9];
  const float* Wk     = (const float*)d_in[10];
  const float* Wv     = (const float*)d_in[11];
  const float* Wr     = (const float*)d_in[12];
  const float* Wo     = (const float*)d_in[13];
  const float* decay  = (const float*)d_in[14];
  const float* first  = (const float*)d_in[15];
  const float* kn_w   = (const float*)d_in[16];
  const float* kn_b   = (const float*)d_in[17];
  const float* sp_w   = (const float*)d_in[18];
  const float* sp_b   = (const float*)d_in[19];
  const float* c1_w   = (const float*)d_in[20];
  const float* c1_b   = (const float*)d_in[21];
  const float* bn1_g  = (const float*)d_in[22];
  const float* bn1_b  = (const float*)d_in[23];
  const float* c2_w   = (const float*)d_in[24];
  const float* c2_b   = (const float*)d_in[25];
  const float* bn2_g  = (const float*)d_in[26];
  const float* bn2_b  = (const float*)d_in[27];
  const float* c3_w   = (const float*)d_in[28];
  const float* c3_b   = (const float*)d_in[29];
  const float* bn3_g  = (const float*)d_in[30];
  const float* bn3_b  = (const float*)d_in[31];
  const float* up_w   = (const float*)d_in[32];
  const float* up_b   = (const float*)d_in[33];
  float* out = (float*)d_out;

  char* base = (char*)d_ws;
  size_t off = 0;
  auto take = [&](size_t bytes) -> char* {
    off = (off + 255) & ~(size_t)255;
    char* p = base + off;
    off += bytes;
    return p;
  };
  const size_t F32 = (size_t)8 * 256 * 4096 * 4;   // 33.5 MB (B,C,H,W) f32
  const size_t F16 = F32 / 2;

  float*    xn     = (float*)take(F32);
  _Float16* xk16   = (_Float16*)take(F16);             // reused later as y3f16
  _Float16* xv16   = (_Float16*)take(F16);
  _Float16* xr16   = (_Float16*)take(F16);
  float*    kbuf   = (float*)take(F32);                // kbuf+vbuf reused as y2 (f16, 1024ch)
  float*    vbuf   = (float*)take(F32);
  float*    srbuf  = (float*)take(F32);                // reused later as y3 (f32)
  float*    wkvh   = (float*)take(F32);                // reused later as y1 (f16, 512ch)
  float*    wkvv   = (float*)take(F32);
  _Float16* srw16  = (_Float16*)take(F16);
  _Float16* xcat   = (_Float16*)take((size_t)8 * 512 * 4096 * 2);
  _Float16* skip16 = (_Float16*)take((size_t)8 * 512 * 4096 * 2);
  _Float16* wk16   = (_Float16*)take(256 * 256 * 2);
  _Float16* wv16   = (_Float16*)take(256 * 256 * 2);
  _Float16* wr16   = (_Float16*)take(256 * 256 * 2);
  _Float16* wo16   = (_Float16*)take(256 * 256 * 2);
  _Float16* sp16   = (_Float16*)take(256 * 512 * 2);
  _Float16* c1w16  = (_Float16*)take(512 * 2304 * 2);
  _Float16* c2w16  = (_Float16*)take(1024 * 512 * 2);
  _Float16* c3w16  = (_Float16*)take(256 * 1024 * 2);
  _Float16* upw16  = (_Float16*)take(512 * 256 * 2);

  _Float16* y1   = (_Float16*)wkvh;
  _Float16* y2   = (_Float16*)kbuf;
  float*    y3   = srbuf;
  _Float16* y316 = xk16;

  const dim3 blk(256);
  auto cdiv = [](long a, long b) { return (unsigned)((a + b - 1) / b); };

  // ---- weight conversion (f32 -> f16) ----
  f32_to_f16_kernel<<<cdiv(65536, 256),  blk, 0, stream>>>(Wk, wk16, 65536);
  f32_to_f16_kernel<<<cdiv(65536, 256),  blk, 0, stream>>>(Wv, wv16, 65536);
  f32_to_f16_kernel<<<cdiv(65536, 256),  blk, 0, stream>>>(Wr, wr16, 65536);
  f32_to_f16_kernel<<<cdiv(65536, 256),  blk, 0, stream>>>(Wo, wo16, 65536);
  f32_to_f16_kernel<<<cdiv(131072, 256), blk, 0, stream>>>(sp_w, sp16, 131072);
  f32_to_f16_kernel<<<cdiv(524288, 256), blk, 0, stream>>>(c2_w, c2w16, 524288);
  f32_to_f16_kernel<<<cdiv(262144, 256), blk, 0, stream>>>(c3_w, c3w16, 262144);
  f32_to_f16_kernel<<<cdiv(131072, 256), blk, 0, stream>>>(up_w, upw16, 131072);
  c1_perm_kernel<<<cdiv(512L * 2304, 256), blk, 0, stream>>>(c1_w, c1w16);
  f32_to_f16_kernel<<<cdiv(16777216, 256), blk, 0, stream>>>(skip, skip16, 16777216);

  // ---- LN1 ----
  ln_chan_kernel<0><<<128, blk, 0, stream>>>(x, ln1_w, ln1_b, xn, nullptr);
  // ---- q_shift + mixes ----
  qshift_mix_kernel<<<32768, blk, 0, stream>>>(xn, mask, mix_k, mix_v, mix_r, xk16, xv16, xr16);

  // ---- K / V / R projections (WMMA) ----
  const dim3 g256(32, 2, 8), g512(32, 4, 8), g1024(32, 8, 8);
  gemm_wmma_kernel<0, EPI_F32><<<g256, blk, 0, stream>>>(wk16, xk16, kbuf, nullptr,
      nullptr, nullptr, nullptr, nullptr, 256, 256L*4096, 256L*4096, 0, 0);
  gemm_wmma_kernel<0, EPI_F32><<<g256, blk, 0, stream>>>(wv16, xv16, vbuf, nullptr,
      nullptr, nullptr, nullptr, nullptr, 256, 256L*4096, 256L*4096, 0, 0);
  gemm_wmma_kernel<0, EPI_SIG><<<g256, blk, 0, stream>>>(wr16, xr16, srbuf, nullptr,
      nullptr, nullptr, nullptr, nullptr, 256, 256L*4096, 256L*4096, 0, 0);

  // ---- directional WKV scans ----
  wkv_hscan_kernel<<<512, blk, 0, stream>>>(kbuf, vbuf, decay, first, wkvh);
  wkv_vscan_kernel<<<512, blk, 0, stream>>>(kbuf, vbuf, decay, first, wkvv);

  // ---- combine + LN(kn) + gate by sigmoid(r) ----
  combine_kernel<<<128, blk, 0, stream>>>(wkvh, wkvv, srbuf, kn_w, kn_b, srw16);

  // ---- output projection + residual -> xcat[0:256] ----
  gemm_wmma_kernel<0, EPI_RESID_F16><<<g256, blk, 0, stream>>>(wo16, srw16, nullptr, xcat,
      nullptr, nullptr, nullptr, xn, 256, 256L*4096, 512L*4096, 256L*4096, 0);
  // ---- skip projection -> xcat[256:512] ----
  gemm_wmma_kernel<0, EPI_BIAS_F16><<<g256, blk, 0, stream>>>(sp16, skip16, nullptr, xcat,
      sp_b, nullptr, nullptr, nullptr, 512, 512L*4096, 512L*4096, 0, 256);

  // ---- grouped 3x3 conv as implicit GEMM (K = 9*256) + GELU + BN ----
  gemm_wmma_kernel<1, EPI_GBN_F16><<<g512, blk, 0, stream>>>(c1w16, xcat, nullptr, y1,
      c1_b, bn1_g, bn1_b, nullptr, 2304, 512L*4096, 512L*4096, 0, 0);
  // ---- c2 1x1 (512->1024) + GELU + BN ----
  gemm_wmma_kernel<0, EPI_GBN_F16><<<g1024, blk, 0, stream>>>(c2w16, y1, nullptr, y2,
      c2_b, bn2_g, bn2_b, nullptr, 512, 512L*4096, 1024L*4096, 0, 0);
  // ---- c3 1x1 (1024->256) + GELU + BN -> f32 for LN2 ----
  gemm_wmma_kernel<0, EPI_GBN_F32><<<g256, blk, 0, stream>>>(c3w16, y2, y3, nullptr,
      c3_b, bn3_g, bn3_b, nullptr, 1024, 1024L*4096, 256L*4096, 0, 0);

  // ---- LN2 -> f16 ----
  ln_chan_kernel<1><<<128, blk, 0, stream>>>(y3, ln2_w, ln2_b, nullptr, y316);

  // ---- patch-expand 1x1 (256->512) + pixel shuffle -> d_out ----
  gemm_wmma_kernel<0, EPI_UP><<<g512, blk, 0, stream>>>(upw16, y316, out, nullptr,
      up_b, nullptr, nullptr, nullptr, 256, 256L*4096, 0, 0, 0);
}
